// LIFRNN_79242146611231
// MI455X (gfx1250) — matrix operand order repaired and checked by reading, not compiled, not run on hardware
//
#include <hip/hip_runtime.h>
#include <stdint.h>

// ---------------------------------------------------------------------------
// LIF RNN for MI455X (gfx1250): bf16 WMMA input-projection GEMM + persistent
// cooperative scan kernel with TDM spike broadcast and WMMA recurrence.
// ---------------------------------------------------------------------------

typedef __attribute__((ext_vector_type(16))) __bf16       v16bf;
typedef __attribute__((ext_vector_type(8)))  float        v8f;
typedef __attribute__((ext_vector_type(4)))  unsigned int v4u;
typedef __attribute__((ext_vector_type(8)))  int          v8i;
typedef __attribute__((ext_vector_type(4)))  int          v4i;

#define LIF_DECAY  0.3f
#define LIF_THRESH 0.5f

constexpr int B = 32, I = 1024, H = 1024, T = 500;
constexpr int TT = 16;                      // t-tile for xs GEMM

// Padded LDS row: 1024 bf16 (2048B) + 16B pad after each 512 halfs (256 dw)
// -> stride 2080B, 8-bank stagger per row. Matches TDM pad_interval=256dw,
// pad_amount=4dw so tensor_load_to_lds can write the padded layout directly.
constexpr unsigned ROWB    = 2080;
constexpr unsigned WH_OFF  = 0;
constexpr unsigned SPK_OFF = 32u * ROWB;            // 66560
constexpr unsigned LDS_BYTES = SPK_OFF + 32u * ROWB; // 133120

__device__ __forceinline__ unsigned lofs(unsigned koff) {
  return koff * 2u + (koff >> 9) * 16u;   // byte offset inside a padded row
}

#if defined(__HIP_DEVICE_COMPILE__) && \
    __has_builtin(__builtin_amdgcn_tensor_load_to_lds) && \
    __has_builtin(__builtin_amdgcn_s_wait_tensorcnt)
#define USE_TDM 1
#else
#define USE_TDM 0
#endif

// ---------------------------------------------------------------------------
// Kernel 1: xs[b][h][t] = sum_i Wx[h,i] * data[b,i,t]  + (bias added at store)
// Per-b GEMM: A = Wx tile [16h x 32k] (K contiguous), B = data tile
// [32k x 16t] (N=t contiguous, K strided by T). One 16x16 D tile per wave.
// ---------------------------------------------------------------------------
__global__ void __launch_bounds__(256)
xs_gemm_kernel(const float* __restrict__ data, const float* __restrict__ Wx,
               const float* __restrict__ bx, float* __restrict__ xs)
{
  const int lane = threadIdx.x & 31;
  const int wave = threadIdx.x >> 5;
  const int lm   = lane & 15;
  const int half = lane >> 4;
  const int t0   = blockIdx.x * TT;
  const int h0   = blockIdx.y * 128 + wave * 16;
  const int b    = blockIdx.z;

  const int tl = (t0 + lm < T) ? (t0 + lm) : (T - 1);   // clamp loads
  const float* arow = Wx + (size_t)(h0 + lm) * I;       // A row (M=lane)
  const float* bcol = data + (size_t)b * I * T + tl;    // B col (N=lane%16)

  v8f c = {};
  #pragma unroll 4
  for (int kk = 0; kk < I / 32; ++kk) {
    v16bf a, bb;
    // A 16x32 bf16: lane-half 0 -> K 0..15, half 1 -> K 16..31
    const float* ap = arow + kk * 32 + half * 16;
    #pragma unroll
    for (int i = 0; i < 8; ++i) {
      a[i]     = (__bf16)ap[i];
      a[i + 8] = (__bf16)ap[8 + i];
    }
    // B 32x16 bf16: lanes 0-15 K=0..15, lanes 16-31 K=16..31 (stride T)
    const float* bp = bcol + (size_t)(kk * 32 + half * 16) * T;
    #pragma unroll
    for (int i = 0; i < 16; ++i) bb[i] = (__bf16)bp[(size_t)i * T];

    c = __builtin_amdgcn_wmma_f32_16x16x32_bf16(false, a, false, bb,
                                                (short)0, c, false, false);
  }

  const int t = t0 + lm;
  if (t < T) {
    #pragma unroll
    for (int j = 0; j < 8; ++j) {
      int h = h0 + j + 8 * half;                  // C: M = j + 8*half
      xs[((size_t)b * H + h) * T + t] = c[j] + bx[h];
    }
  }
}

// ---------------------------------------------------------------------------
// Kernel 2: persistent scan. 32 blocks x 128 threads (4 waves); block owns
// h-slice [h0, h0+32). Wave: btile = wave&1 (b rows), htile = wave>>1.
// ---------------------------------------------------------------------------
__global__ void __launch_bounds__(128)
lif_scan_kernel(const float* __restrict__ xs, const float* __restrict__ Wh,
                float* __restrict__ out, __bf16* __restrict__ sb0,
                __bf16* __restrict__ sb1, unsigned* __restrict__ cnt)
{
  extern __shared__ char smem[];
  char* wh_p  = smem + WH_OFF;    // Wh slice  [32h x 1024k] bf16, padded rows
  char* spk_p = smem + SPK_OFF;   // spk_prev  [32b x 1024h] bf16, padded rows

  const int tid   = threadIdx.x;
  const int lane  = tid & 31;
  const int wave  = tid >> 5;
  const int lm    = lane & 15;
  const int half  = lane >> 4;
  const int btile = wave & 1;
  const int htile = (wave >> 1) & 1;
  const int h0    = blockIdx.x * 32;
  const int nb    = gridDim.x;

  // Stage Wh rows [h0, h0+32) as bf16 into LDS once (reused 500x).
  for (int i = tid; i < 32 * 1024; i += blockDim.x) {
    int r = i >> 10, k = i & 1023;
    *(__bf16*)(wh_p + (size_t)r * ROWB + lofs((unsigned)k)) =
        (__bf16)Wh[(size_t)(h0 + r) * H + k];
  }
  __syncthreads();

  float mem[8], spk[8];
  #pragma unroll
  for (int j = 0; j < 8; ++j) { mem[j] = 0.f; spk[j] = 0.f; }

  const int habs = h0 + htile * 16 + lm;          // C/D: N = lane%16

  for (int t = 0; t < T; ++t) {
    const __bf16* scur = (t & 1) ? sb1 : sb0;
    __bf16*       snxt = (t & 1) ? sb0 : sb1;

    // ---- broadcast spk_prev [32 x 1024] bf16 into LDS ----
#if USE_TDM
    if (wave == 0) {
      typedef __attribute__((address_space(3))) char lds_char;
      unsigned lds_dst = (unsigned)(uintptr_t)(lds_char*)spk_p;
      unsigned long long ga = (unsigned long long)(uintptr_t)scur;
      // D# group0: count=1 | lds_addr | global_addr[56:0] | type=2
      v4u g0 = { 1u, lds_dst, (unsigned)(ga & 0xFFFFFFFFu),
                 (unsigned)(((ga >> 32) & 0x1FFFFFFu) | (2u << 30)) };
      // D# group1: data_size=2B, pad_enable, pad_interval=256dw, pad_amount=4dw
      v8i g1 = {};
      g1[0] = (int)((1u << 16) | (1u << 20) | (7u << 22) | (3u << 25));
      g1[1] = (int)(1024u << 16);   // tensor_dim0 = 1024 (lo16 @ bits63:48)
      g1[2] = (int)(32u << 16);     // dim0 hi16=0 | tensor_dim1=32 lo16
      g1[3] = (int)(1024u << 16);   // dim1 hi16=0 | tile_dim0 = 1024
      g1[4] = (int)32u;             // tile_dim1 = 32, tile_dim2 = 0
      g1[5] = (int)1024u;           // tensor_dim0_stride = 1024
      g1[6] = 0; g1[7] = 0;
      v4i z4 = {0, 0, 0, 0};
      v8i z8 = {};
      __builtin_amdgcn_tensor_load_to_lds(g0, g1, z4, z4, z8, 0);
      __builtin_amdgcn_s_wait_tensorcnt(0);
    }
#else
    for (int c2 = tid; c2 < 32 * 128; c2 += blockDim.x) {   // 16B chunks
      int r = c2 >> 7;
      unsigned koff = (unsigned)(c2 & 127) * 8u;
      uint4 v = *(const uint4*)((const char*)scur + (size_t)r * 2048 + koff * 2);
      *(uint4*)(spk_p + (size_t)r * ROWB + lofs(koff)) = v;
    }
#endif
    __syncthreads();

    // Seed C with leak/reset + input drive: mem*DECAY*(1-spk) + x_t
    v8f c;
    #pragma unroll
    for (int j = 0; j < 8; ++j) {
      int b = btile * 16 + j + 8 * half;
      float x = xs[((size_t)b * H + habs) * T + t];
      c[j] = mem[j] * LIF_DECAY * (1.0f - spk[j]) + x;
    }
    if (t + 16 < T) {   // global_prefetch_b8 for a future xs line
      int b0 = btile * 16 + 8 * half;
      __builtin_prefetch(&xs[((size_t)b0 * H + habs) * T + t + 16], 0, 1);
    }

    // Recurrent GEMM: D += spk_prev[16b x 1024] * Wh^T[1024 x 16h]
    const char* arow_p = spk_p + (size_t)(btile * 16 + lm) * ROWB;
    const char* brow_p = wh_p  + (size_t)(htile * 16 + lm) * ROWB;
    #pragma unroll 8
    for (int kk = 0; kk < 32; ++kk) {
      union { v16bf v; uint4 u[2]; } a, bm;
      unsigned ka = (unsigned)kk * 32u + (unsigned)half * 16u;
      a.u[0] = *(const uint4*)(arow_p + lofs(ka));
      a.u[1] = *(const uint4*)(arow_p + lofs(ka + 8u));
      unsigned kb = (unsigned)kk * 32u + (unsigned)half * 16u;
      bm.u[0] = *(const uint4*)(brow_p + lofs(kb));
      bm.u[1] = *(const uint4*)(brow_p + lofs(kb) + 16u);
      c = __builtin_amdgcn_wmma_f32_16x16x32_bf16(false, a.v, false, bm.v,
                                                  (short)0, c, false, false);
    }

    // Threshold, emit spikes (out + next-step broadcast buffer), update state.
    #pragma unroll
    for (int j = 0; j < 8; ++j) {
      int b = btile * 16 + j + 8 * half;
      mem[j] = c[j];
      float s = (mem[j] - LIF_THRESH > 0.f) ? 1.f : 0.f;
      spk[j] = s;
      out[((size_t)b * H + habs) * T + t] = s;
      snxt[(size_t)b * H + habs] = (__bf16)s;
    }

    // ---- grid-wide step barrier ----
    __threadfence();
    __syncthreads();
    if (tid == 0) {
      __hip_atomic_fetch_add(cnt, 1u, __ATOMIC_RELEASE, __HIP_MEMORY_SCOPE_AGENT);
      unsigned target = (unsigned)(nb * (t + 1));
      while (__hip_atomic_load(cnt, __ATOMIC_ACQUIRE, __HIP_MEMORY_SCOPE_AGENT)
             < target)
        __builtin_amdgcn_s_sleep(1);
    }
    __syncthreads();
  }
}

// ---------------------------------------------------------------------------
// Init: zero spike double-buffers and the barrier counter (every launch, so
// graph replays are deterministic).
// ---------------------------------------------------------------------------
__global__ void lif_init_kernel(__bf16* sb0, __bf16* sb1, unsigned* cnt)
{
  int i = blockIdx.x * blockDim.x + threadIdx.x;
  int stride = gridDim.x * blockDim.x;
  for (int k = i; k < B * H; k += stride) {
    sb0[k] = (__bf16)0.f;
    sb1[k] = (__bf16)0.f;
  }
  if (i == 0) *cnt = 0u;
}

// ---------------------------------------------------------------------------
extern "C" void kernel_launch(void* const* d_in, const int* in_sizes, int n_in,
                              void* d_out, int out_size, void* d_ws, size_t ws_size,
                              hipStream_t stream)
{
  const float* data = (const float*)d_in[0];   // [B, I, T]
  const float* Wx   = (const float*)d_in[1];   // [H, I]
  const float* bx   = (const float*)d_in[2];   // [H]
  const float* Wh   = (const float*)d_in[3];   // [H, H]
  float* out = (float*)d_out;                  // [B, H, T]

  char* ws = (char*)d_ws;
  const size_t XS_BYTES = (size_t)B * H * T * sizeof(float);   // ~65.5 MB
  float*  xs  = (float*)ws;
  __bf16* sb0 = (__bf16*)(ws + XS_BYTES);
  __bf16* sb1 = sb0 + (size_t)B * H;
  unsigned* cnt = (unsigned*)(ws + XS_BYTES + 2 * (size_t)B * H * sizeof(__bf16));

  lif_init_kernel<<<dim3(64), 256, 0, stream>>>(sb0, sb1, cnt);

  dim3 g1((T + TT - 1) / TT, H / 128, B);      // 32 x 8 x 32 blocks
  xs_gemm_kernel<<<g1, 256, 0, stream>>>(data, Wx, bx, xs);

  lif_scan_kernel<<<dim3(H / 32), 128, LDS_BYTES, stream>>>(xs, Wh, out,
                                                            sb0, sb1, cnt);
}
